// TalkingHeadAttention_15685220565608
// MI455X (gfx1250) — matrix operand order repaired and checked by reading, not compiled, not run on hardware
//
#include <hip/hip_runtime.h>
#include <hip/hip_bf16.h>
#include <cstdint>
#include <cstddef>

// ---------------------------------------------------------------------------
// Types for CDNA5 WMMA (wave32): D(16x16 f32) = A(16x32 bf16) x B(32x16 bf16) + C
// ---------------------------------------------------------------------------
typedef __attribute__((ext_vector_type(16))) __bf16 v16bf;
typedef __attribute__((ext_vector_type(8)))  float  v8f;

union BF16Frag {
    v16bf    v;
    __bf16   e[16];
    uint32_t u[8];
};

#define WMMA_BF16(A, B, C) \
    __builtin_amdgcn_wmma_f32_16x16x32_bf16(false, (A), false, (B), (short)0, (C), false, false)

// A-fragment (16x32 bf16, lane = M row in both halves):
//   lane<16 : K {0..15};  lane>=16 : K {16..31}
// uint (bf16-pair) index for vgpr i, given half = lane>>4:
//   -> one contiguous group of 8 uints => compiler emits 2x global_load_b128
__device__ __forceinline__ int a_uidx(int i, int half) {
    return half * 8 + i;
}

// ---------------------------------------------------------------------------
// Kernel 1: f32 -> bf16 conversion (vectorized: float4 in, 4x bf16 out)
// ---------------------------------------------------------------------------
union PK2 { __bf16 h[2]; uint32_t u; };

__global__ void k_f32_to_bf16(const float4* __restrict__ in, uint2* __restrict__ out, int n4) {
    int i = blockIdx.x * blockDim.x + threadIdx.x;
    int s = gridDim.x * blockDim.x;
    for (; i < n4; i += s) {
        float4 f = in[i];
        PK2 lo, hi;
        lo.h[0] = (__bf16)f.x; lo.h[1] = (__bf16)f.y;
        hi.h[0] = (__bf16)f.z; hi.h[1] = (__bf16)f.w;
        out[i] = make_uint2(lo.u, hi.u);
    }
}

// ---------------------------------------------------------------------------
// Kernel 2: WMMA GEMM   C[M,N] = A[M,K](bf16) * W[N,K]^T(bf16) + bias[N]
// MODE 0: store bf16 [M][N]
// MODE 1: store bf16 per-batch transposed: out[(b*1024+col)*1024 + (m&1023)]
// MODE 2: store f32  [M][N]
// block = 256 thr (8 waves as 4x2), wave does 4x2 16-tiles => block tile 256x64
// per k-step: 12 b128 loads, 8 WMMA (B frags reused across 4 A rows)
// ---------------------------------------------------------------------------
template <int MODE>
__global__ __launch_bounds__(256)
void k_gemm_bf16(const __bf16* __restrict__ A, const __bf16* __restrict__ W,
                 const float* __restrict__ bias, void* __restrict__ out,
                 int M, int N, int K) {
    const int tid  = threadIdx.x;
    const int lane = tid & 31;
    const int wv   = tid >> 5;
    const int half = lane >> 4;
    const int ln   = lane & 15;

    const int m_w = blockIdx.y * 256 + (wv & 3) * 64;   // 4 M-tiles of 16
    const int n_w = blockIdx.x * 64  + (wv >> 2) * 32;  // 2 N-tiles of 16

    v8f acc[4][2] = {};

    const uint32_t* arow[4];
#pragma unroll
    for (int tm = 0; tm < 4; ++tm)
        arow[tm] = (const uint32_t*)(A + (size_t)(m_w + tm * 16 + ln) * K);
    const uint32_t* brow0 = (const uint32_t*)(W + (size_t)(n_w + ln) * K);
    const uint32_t* brow1 = (const uint32_t*)(W + (size_t)(n_w + 16 + ln) * K);

    for (int kc = 0; kc < K; kc += 32) {
        const int kcu = kc >> 1;
        BF16Frag a[4], b0, b1;
#pragma unroll
        for (int tm = 0; tm < 4; ++tm)
#pragma unroll
            for (int i = 0; i < 8; ++i)
                a[tm].u[i] = arow[tm][kcu + a_uidx(i, half)];
        // B fragment: lane = N col, K = half*16 + e (16 contiguous bf16 = 8 uints)
        const int bu = kcu + half * 8;
#pragma unroll
        for (int i = 0; i < 8; ++i) {
            b0.u[i] = brow0[bu + i];
            b1.u[i] = brow1[bu + i];
        }
#pragma unroll
        for (int tm = 0; tm < 4; ++tm) {
            acc[tm][0] = WMMA_BF16(a[tm].v, b0.v, acc[tm][0]);
            acc[tm][1] = WMMA_BF16(a[tm].v, b1.v, acc[tm][1]);
        }
    }

#pragma unroll
    for (int tm = 0; tm < 4; ++tm) {
#pragma unroll
        for (int tn = 0; tn < 2; ++tn) {
            const int col = n_w + tn * 16 + ln;
            const float bcol = bias[col];
#pragma unroll
            for (int r = 0; r < 8; ++r) {
                const int m = m_w + tm * 16 + r + half * 8;
                const float val = acc[tm][tn][r] + bcol;
                if (MODE == 0) {
                    ((__bf16*)out)[(size_t)m * N + col] = (__bf16)val;
                } else if (MODE == 1) {
                    const int b = m >> 10, ql = m & 1023;
                    ((__bf16*)out)[((size_t)(b * 1024 + col)) * 1024 + ql] = (__bf16)val;
                } else {
                    ((float*)out)[(size_t)m * N + col] = val;
                }
            }
        }
    }
}

// ---------------------------------------------------------------------------
// Kernel 3: addv[b*1024+col] = bw[col>>6] * sum_k Vt[b][col][k]
// (analytic handling of the post-softmax constant weight bias bw)
// one wave per row, 8 waves/block
// ---------------------------------------------------------------------------
__global__ void k_addv(const __bf16* __restrict__ Vt, const float* __restrict__ bw,
                       float* __restrict__ addv) {
    const int row  = blockIdx.x * 8 + (threadIdx.x >> 5);
    const int lane = threadIdx.x & 31;
    const __bf16* p = Vt + (size_t)row * 1024;
    float s = 0.f;
    for (int k = lane; k < 1024; k += 32) s += (float)p[k];
#pragma unroll
    for (int m = 16; m; m >>= 1) s += __shfl_xor(s, m, 32);
    if (lane == 0) addv[row] = s * bw[(row & 1023) >> 6];
}

// ---------------------------------------------------------------------------
// Kernel 4: talking-heads attention core.
// grid = (64 q-tiles, 4 batches); block = 512 thr = 16 waves, wave = head = g.
// Two-pass online softmax over Lk with cross-wave head mixing via LDS.
// ---------------------------------------------------------------------------
__global__ __launch_bounds__(512)
void k_attn(const __bf16* __restrict__ Qp,         // [B*1024][1024]
            const __bf16* __restrict__ Kp,         // [B*1024][1024]
            const __bf16* __restrict__ Vt,         // [b][col][k] bf16
            const float*  __restrict__ attn_bias,  // [B][16][1024]
            const float*  __restrict__ Wl, const float* __restrict__ bl,
            const float*  __restrict__ Ww,
            const float*  __restrict__ addv,       // [B][1024]
            __bf16*       __restrict__ ctx)        // [B*1024][1024] bf16
{
    __shared__ float  sc[8][32][17];        // stride 17 -> conflict-free (17 coprime 64)
    __shared__ __bf16 wlds[16][16][32];     // per-wave C-layout -> A-layout staging

    const int tid  = threadIdx.x;
    const int lane = tid & 31;
    const int h    = tid >> 5;              // wave index == head index == mixed-head g
    const int half = lane >> 4;
    const int ln   = lane & 15;

    const int q0 = blockIdx.x * 16;
    const int b  = blockIdx.y;

    // per-wave mixing rows
    float wl[16], ww[16];
#pragma unroll
    for (int j = 0; j < 16; ++j) { wl[j] = Wl[h * 16 + j]; ww[j] = Ww[h * 16 + j]; }

    // pre-mixed per-row bias: bmix = bl[g] + sum_h Wl[g,h]*attn_bias[b,h,q]
    float bmix[8];
#pragma unroll
    for (int r = 0; r < 8; ++r) {
        const int qrow = q0 + r + half * 8;
        float acc = bl[h];
#pragma unroll
        for (int j = 0; j < 16; ++j)
            acc += wl[j] * attn_bias[(((size_t)b * 16 + j) << 10) + qrow];
        bmix[r] = acc;
    }

    // Q A-fragments for this head: 2 chunks of K=32 over d=64 (kept in VGPRs)
    BF16Frag aq[2];
    {
        const uint32_t* qrow =
            (const uint32_t*)(Qp + ((size_t)(b * 1024 + q0 + ln)) * 1024 + h * 64);
#pragma unroll
        for (int c = 0; c < 2; ++c)
#pragma unroll
            for (int i = 0; i < 8; ++i)
                aq[c].u[i] = qrow[c * 16 + a_uidx(i, half)];
    }

    // logits for one 16-wide k-tile (raw scores via WMMA + cross-wave Wl mix)
    auto compute_logits = [&](int k0, float* lout) {
        BF16Frag bk0, bk1;
        const uint32_t* kr =
            (const uint32_t*)(Kp + ((size_t)(b * 1024 + k0 + ln)) * 1024 + h * 64);
#pragma unroll
        for (int i = 0; i < 8; ++i) {
            bk0.u[i] = kr[half * 8 + i];          // d-chunk 0
            bk1.u[i] = kr[16 + half * 8 + i];     // d-chunk 1
        }
        v8f s = {};
        s = WMMA_BF16(aq[0].v, bk0.v, s);
        s = WMMA_BF16(aq[1].v, bk1.v, s);
        __syncthreads();                          // previous readers done
#pragma unroll
        for (int r = 0; r < 8; ++r) sc[r][lane][h] = s[r] * 0.125f;   // DIM_KEY^-0.5
        __syncthreads();
#pragma unroll
        for (int r = 0; r < 8; ++r) {
            float acc = bmix[r];
#pragma unroll
            for (int j = 0; j < 16; ++j) acc += wl[j] * sc[r][lane][j];
            lout[r] = acc;
        }
    };

    // ---------------- pass 1: online softmax statistics over Lk -------------
    float m_l[8], s_l[8];
#pragma unroll
    for (int r = 0; r < 8; ++r) { m_l[r] = -1e30f; s_l[r] = 0.f; }

    for (int k0 = 0; k0 < 1024; k0 += 16) {
        if (k0 + 16 < 1024)
            __builtin_prefetch(Kp + ((size_t)(b * 1024 + k0 + 16 + ln)) * 1024 + h * 64, 0, 3);
        float l[8];
        compute_logits(k0, l);
#pragma unroll
        for (int r = 0; r < 8; ++r) {
            const float nm = fmaxf(m_l[r], l[r]);
            s_l[r] = s_l[r] * __expf(m_l[r] - nm) + __expf(l[r] - nm);
            m_l[r] = nm;
        }
    }
    // combine across the 16 lanes (same rows) of each half
#pragma unroll
    for (int mask = 1; mask <= 8; mask <<= 1) {
#pragma unroll
        for (int r = 0; r < 8; ++r) {
            const float om = __shfl_xor(m_l[r], mask, 32);
            const float os = __shfl_xor(s_l[r], mask, 32);
            const float nm = fmaxf(m_l[r], om);
            s_l[r] = s_l[r] * __expf(m_l[r] - nm) + os * __expf(om - nm);
            m_l[r] = nm;
        }
    }
    float rinv[8];
#pragma unroll
    for (int r = 0; r < 8; ++r) rinv[r] = 1.0f / s_l[r];

    // ---------------- pass 2: normalize, Ww mix, ctx = W * V -----------------
    v8f cacc[4] = {};
    const __bf16* vbase = Vt + ((size_t)(b * 1024 + h * 64 + ln)) * 1024;

    for (int kt = 0; kt < 1024; kt += 32) {
#pragma unroll
        for (int s2 = 0; s2 < 2; ++s2) {
            const int k0 = kt + s2 * 16;
            float l[8];
            compute_logits(k0, l);
            float p[8];
#pragma unroll
            for (int r = 0; r < 8; ++r) p[r] = __expf(l[r] - m_l[r]) * rinv[r];
            __syncthreads();
#pragma unroll
            for (int r = 0; r < 8; ++r) sc[r][lane][h] = p[r];
            __syncthreads();
#pragma unroll
            for (int r = 0; r < 8; ++r) {
                float acc = 0.f;
#pragma unroll
                for (int j = 0; j < 16; ++j) acc += ww[j] * sc[r][lane][j];
                // C-layout element (m = r+8*half, n = ln) -> A-layout staging
                wlds[h][r + half * 8][s2 * 16 + ln] = (__bf16)acc;
            }
        }
        // A fragment (16x32) for this wave from its private staging row
        BF16Frag aw;
        const uint32_t* wrow = (const uint32_t*)(&wlds[h][ln][0]);
#pragma unroll
        for (int i = 0; i < 8; ++i) aw.u[i] = wrow[a_uidx(i, half)];

        // V B-fragments: 4 d-chunks of 16; Vt is K-contiguous -> 8 uints each
#pragma unroll
        for (int nc = 0; nc < 4; ++nc) {
            BF16Frag bv;
            const uint32_t* vp =
                (const uint32_t*)(vbase + (size_t)nc * 16 * 1024 + kt + half * 16);
#pragma unroll
            for (int i = 0; i < 8; ++i) bv.u[i] = vp[i];
            cacc[nc] = WMMA_BF16(aw.v, bv.v, cacc[nc]);
        }
    }

    // store ctx (+ analytic bw * sum_k v term)
#pragma unroll
    for (int nc = 0; nc < 4; ++nc) {
        const int col = h * 64 + nc * 16 + ln;
        const float av = addv[b * 1024 + col];
#pragma unroll
        for (int r = 0; r < 8; ++r) {
            const int m = q0 + r + half * 8;
            ctx[((size_t)(b * 1024 + m)) * 1024 + col] = (__bf16)(cacc[nc][r] + av);
        }
    }
}

// ---------------------------------------------------------------------------
// Host launcher
// ---------------------------------------------------------------------------
extern "C" void kernel_launch(void* const* d_in, const int* in_sizes, int n_in,
                              void* d_out, int out_size, void* d_ws, size_t ws_size,
                              hipStream_t stream) {
    (void)in_sizes; (void)n_in; (void)out_size; (void)ws_size;

    const float* queries   = (const float*)d_in[0];
    const float* keys      = (const float*)d_in[1];
    const float* values    = (const float*)d_in[2];
    /* d_in[3] = mask (unused by the module) */
    const float* attn_bias = (const float*)d_in[4];
    const float* Wq = (const float*)d_in[5];  const float* bq = (const float*)d_in[6];
    const float* Wk = (const float*)d_in[7];  const float* bk = (const float*)d_in[8];
    const float* Wv = (const float*)d_in[9];  const float* bv = (const float*)d_in[10];
    const float* Wl = (const float*)d_in[11]; const float* bl = (const float*)d_in[12];
    const float* Ww = (const float*)d_in[13]; const float* bw = (const float*)d_in[14];
    const float* Wo = (const float*)d_in[15]; const float* bo = (const float*)d_in[16];

    const int nAct = 4096 * 1024;   // B*L x DIM_MODEL
    const int nW   = 1024 * 1024;

    char* ws = (char*)d_ws;
    size_t off = 0;
    auto carve = [&](size_t bytes) -> char* {
        char* p = ws + off;
        off += (bytes + 255) & ~(size_t)255;
        return p;
    };
    __bf16* xq   = (__bf16*)carve((size_t)nAct * 2);
    __bf16* xk   = (__bf16*)carve((size_t)nAct * 2);
    __bf16* xv   = (__bf16*)carve((size_t)nAct * 2);
    __bf16* wqb  = (__bf16*)carve((size_t)nW * 2);
    __bf16* wkb  = (__bf16*)carve((size_t)nW * 2);
    __bf16* wvb  = (__bf16*)carve((size_t)nW * 2);
    __bf16* wob  = (__bf16*)carve((size_t)nW * 2);
    __bf16* Qp   = (__bf16*)carve((size_t)nAct * 2);
    __bf16* Kp   = (__bf16*)carve((size_t)nAct * 2);
    __bf16* Vt   = (__bf16*)carve((size_t)nAct * 2);
    __bf16* ctxb = (__bf16*)carve((size_t)nAct * 2);
    float*  addv = (float*) carve((size_t)4 * 1024 * 4);

    // 1) f32 -> bf16 (vectorized)
    k_f32_to_bf16<<<2048, 256, 0, stream>>>((const float4*)queries, (uint2*)xq, nAct / 4);
    k_f32_to_bf16<<<2048, 256, 0, stream>>>((const float4*)keys,    (uint2*)xk, nAct / 4);
    k_f32_to_bf16<<<2048, 256, 0, stream>>>((const float4*)values,  (uint2*)xv, nAct / 4);
    k_f32_to_bf16<<<1024, 256, 0, stream>>>((const float4*)Wq, (uint2*)wqb, nW / 4);
    k_f32_to_bf16<<<1024, 256, 0, stream>>>((const float4*)Wk, (uint2*)wkb, nW / 4);
    k_f32_to_bf16<<<1024, 256, 0, stream>>>((const float4*)Wv, (uint2*)wvb, nW / 4);
    k_f32_to_bf16<<<1024, 256, 0, stream>>>((const float4*)Wo, (uint2*)wob, nW / 4);

    // 2) projections (4096x1024 @ 1024x1024^T), block tile 256x64
    dim3 gg(16, 16), gb(256);
    k_gemm_bf16<0><<<gg, gb, 0, stream>>>(xq, wqb, bq, Qp, 4096, 1024, 1024);
    k_gemm_bf16<0><<<gg, gb, 0, stream>>>(xk, wkb, bk, Kp, 4096, 1024, 1024);
    k_gemm_bf16<1><<<gg, gb, 0, stream>>>(xv, wvb, bv, Vt, 4096, 1024, 1024);  // transposed store

    // 3) bw * sum_k v
    k_addv<<<512, 256, 0, stream>>>(Vt, bw, addv);

    // 4) talking-heads attention core
    dim3 ga(64, 4);
    k_attn<<<ga, 512, 0, stream>>>(Qp, Kp, Vt, attn_bias, Wl, bl, Ww, addv, ctxb);

    // 5) output projection -> f32 d_out
    k_gemm_bf16<2><<<gg, gb, 0, stream>>>(ctxb, wob, bo, d_out, 4096, 1024, 1024);
}
